// ChannelProjection_25211458027700
// MI455X (gfx1250) — matrix operand order repaired and checked by reading, not compiled, not run on hardware
//
#include <hip/hip_runtime.h>
#include <hip/hip_bf16.h>

typedef __attribute__((ext_vector_type(2)))  _Float16 h2;
typedef __attribute__((ext_vector_type(4)))  _Float16 h4;
typedef __attribute__((ext_vector_type(8)))  _Float16 v8h;
typedef __attribute__((ext_vector_type(16))) _Float16 v16h;
typedef __attribute__((ext_vector_type(8)))  float    v8f;
typedef __attribute__((ext_vector_type(4)))  float    f4;   /* native vector: OK for nontemporal builtins */

#define N_    16
#define C_    128
#define CC_   64
#define H_    192
#define W_    192
#define P_    (H_*W_)        /* 36864 pixels per sample            */
#define CPS_  (C_*P_)        /* 4718592 elements per sample        */
#define SBLK_ 288            /* stats blocks per sample            */
#define CHUNK_ (CPS_/SBLK_)  /* 16384 elements per stats block     */
#define PTILE_ 128           /* pixels per main block              */
#define EPS_  1e-5f

/* padded LDS row strides (in halves) to break bank conflicts */
#define Z1RS 72
#define W1RS 72
#define W2RS 136
#define H1RS 136

__device__ inline v16h cat8(v8h a, v8h b){
  return __builtin_shufflevector(a, b, 0,1,2,3,4,5,6,7,8,9,10,11,12,13,14,15);
}

/* A-fragment (16x32 f16): lane row = lane&15; halves 0..7 hold K = kb+8*hi..+7,
   halves 8..15 hold K = kb+16+8*hi..+7  (CDNA5 ISA 7.12.2 16-bit A layout) */
__device__ inline v16h load_a(const _Float16* w, int rs, int row_base, int kb, int lane){
  int r  = row_base + (lane & 15);
  int hi = lane >> 4;
  const _Float16* p = w + r*rs + kb + hi*8;
  return cat8(*(const v8h*)p, *(const v8h*)(p + 16));
}

/* B-fragment (32x16 f16): lane col = lane&15; halves 0..7 hold K = kb+8*hi..+7,
   halves 8..15 hold K = kb+16+8*hi..+7  (same K order as A fragment) */
__device__ inline v16h load_b(const _Float16* act, int rs, int pos_base, int kb, int lane){
  int c  = lane & 15;
  int hi = lane >> 4;
  const _Float16* p = act + (pos_base + c)*rs + kb + hi*8;
  return cat8(*(const v8h*)p, *(const v8h*)(p + 16));
}

/* silu via v_exp_f32 + v_rcp_f32 (avoid IEEE div expansion) */
__device__ inline float silu_f(float x){
  return x * __builtin_amdgcn_rcpf(1.0f + __expf(-x));
}

/* ---------------- pass 1: per-sample partial sums (float4 streams) ---------------- */
__global__ __launch_bounds__(256) void cp_stats1(const float* __restrict__ z0,
                                                 float2* __restrict__ partials){
  __shared__ float ssum[256];
  __shared__ float ssq [256];
  const int n = blockIdx.y, b = blockIdx.x, t = threadIdx.x;
  const float* p = z0 + (size_t)n*CPS_ + (size_t)b*CHUNK_;
  float s = 0.f, q = 0.f;
  #pragma unroll 4
  for (int i = 0; i < CHUNK_/(256*4); ++i){
    f4 v = *(const f4*)(p + (t + i*256)*4);
    s += v.x + v.y + v.z + v.w;
    q += v.x*v.x + v.y*v.y + v.z*v.z + v.w*v.w;
  }
  ssum[t] = s; ssq[t] = q;
  __syncthreads();
  for (int off = 128; off > 0; off >>= 1){
    if (t < off){ ssum[t] += ssum[t+off]; ssq[t] += ssq[t+off]; }
    __syncthreads();
  }
  if (t == 0) partials[n*SBLK_ + b] = make_float2(ssum[0], ssq[0]);
}

/* ---------------- pass 2: finalize mean / rsqrt(var) ---------------- */
__global__ __launch_bounds__(256) void cp_stats2(const float2* __restrict__ partials,
                                                 float2* __restrict__ musig){
  __shared__ float ssum[256];
  __shared__ float ssq [256];
  const int n = blockIdx.x, t = threadIdx.x;
  float s = 0.f, q = 0.f;
  for (int i = t; i < SBLK_; i += 256){
    float2 v = partials[n*SBLK_ + i];
    s += v.x; q += v.y;
  }
  ssum[t] = s; ssq[t] = q;
  __syncthreads();
  for (int off = 128; off > 0; off >>= 1){
    if (t < off){ ssum[t] += ssum[t+off]; ssq[t] += ssq[t+off]; }
    __syncthreads();
  }
  if (t == 0){
    const float inv  = 1.0f / (float)CPS_;
    const float mean = ssum[0] * inv;
    const float var  = ssq[0] * inv - mean*mean;
    musig[n] = make_float2(mean, rsqrtf(var + EPS_));
  }
}

/* ------------- pass 3: fused norm + MLP (WMMA) + shuffle + residual ------------- */
__global__ __launch_bounds__(256, 2) void cp_main(const float* __restrict__ z0,
    const float* __restrict__ w1, const float* __restrict__ b1,
    const float* __restrict__ w2, const float* __restrict__ b2,
    const float2* __restrict__ musig, float* __restrict__ out)
{
  __shared__ __align__(32) _Float16 smem[44544];   /* ~87 KB */
  __shared__ float b1s[128];
  __shared__ float b2s[64];
  _Float16* z1s = smem;             /* 128 pos  x Z1RS  (z1 normalized f16) */
  _Float16* w1s = smem + 9216;      /* 128 out  x W1RS                      */
  _Float16* w2s = smem + 18432;     /*  64 out  x W2RS                      */
  _Float16* h1s = smem + 27136;     /* 8 waves x 16 pos x H1RS              */

  const int t     = threadIdx.x;
  const int n     = blockIdx.y;
  const int pbase = blockIdx.x * PTILE_;
  const int nbase = n * CPS_;
  const float2 ms = musig[n];
  const float mu = ms.x, rs = ms.y;

  /* stage z1 tile (64ch x 128px): float4 loads of two adjacent channels,
     normalize, transpose into [pos][ch] f16 with packed ds_store_b32 */
  #pragma unroll
  for (int i = 0; i < 4; ++i){
    int u  = t + i*256;
    int c2 = u >> 5;               /* channel pair 0..31 -> ch = 2*c2 */
    int p4 = (u & 31) * 4;         /* pixel group of 4                */
    const f4 va = *(const f4*)(z0 + nbase + (2*c2  )*P_ + pbase + p4);
    const f4 vb = *(const f4*)(z0 + nbase + (2*c2+1)*P_ + pbase + p4);
    #pragma unroll
    for (int e = 0; e < 4; ++e){
      h2 pk;
      pk[0] = (_Float16)((va[e] - mu) * rs);
      pk[1] = (_Float16)((vb[e] - mu) * rs);
      *(h2*)(z1s + (p4 + e)*Z1RS + 2*c2) = pk;
    }
  }
  /* stage weights f32->f16: b128 load -> ds_store_b64 */
  #pragma unroll
  for (int i = 0; i < 8; ++i){
    int u = t + i*256;
    int o = u >> 4, k4 = (u & 15) * 4;
    f4 v = *(const f4*)(w1 + o*64 + k4);
    h4 pk = {(_Float16)v.x, (_Float16)v.y, (_Float16)v.z, (_Float16)v.w};
    *(h4*)(w1s + o*W1RS + k4) = pk;
  }
  #pragma unroll
  for (int i = 0; i < 8; ++i){
    int u = t + i*256;
    int o = u >> 5, k4 = (u & 31) * 4;
    f4 v = *(const f4*)(w2 + o*128 + k4);
    h4 pk = {(_Float16)v.x, (_Float16)v.y, (_Float16)v.z, (_Float16)v.w};
    *(h4*)(w2s + o*W2RS + k4) = pk;
  }
  if (t < 128) b1s[t] = b1[t];
  if (t <  64) b2s[t] = b2[t];
  __syncthreads();

  const int wave = t >> 5;
  const int lane = t & 31;
  const int col  = lane & 15;
  const int hi   = lane >> 4;
  _Float16* h1w  = h1s + wave*16*H1RS;   /* per-wave private h1 region */

  /* B operand of GEMM1: this wave's 16 pixels, K = 64 channels in 2 steps */
  v16h bz0 = load_b(z1s, Z1RS, wave*16, 0,  lane);
  v16h bz1 = load_b(z1s, Z1RS, wave*16, 32, lane);

  /* GEMM1: h1(128 x 16px) = silu(w1 * z1 + b1) */
  #pragma unroll
  for (int m = 0; m < 8; ++m){
    v8f acc;
    #pragma unroll
    for (int j = 0; j < 8; ++j) acc[j] = b1s[16*m + 8*hi + j];
    v16h a0 = load_a(w1s, W1RS, 16*m, 0, lane);
    acc = __builtin_amdgcn_wmma_f32_16x16x32_f16(false, a0, false, bz0, (short)0, acc, false, false);
    v16h a1 = load_a(w1s, W1RS, 16*m, 32, lane);
    acc = __builtin_amdgcn_wmma_f32_16x16x32_f16(false, a1, false, bz1, (short)0, acc, false, false);
    #pragma unroll
    for (int j = 0; j < 8; j += 2){
      h2 pk;
      pk[0] = (_Float16)silu_f(acc[j]);
      pk[1] = (_Float16)silu_f(acc[j+1]);
      *(h2*)(h1w + col*H1RS + 16*m + 8*hi + j) = pk;   /* [pos][ch] f16 */
    }
  }

  /* B operand of GEMM2: K = 128 h1 channels in 4 steps (same-wave LDS, DS in-order) */
  v16h bh[4];
  #pragma unroll
  for (int k = 0; k < 4; ++k) bh[k] = load_b(h1w, H1RS, 0, 32*k, lane);

  /* GEMM2 + channel-shuffle (even out channels) + residual */
  #pragma unroll
  for (int m = 0; m < 4; ++m){
    v8f acc;
    #pragma unroll
    for (int j = 0; j < 8; ++j) acc[j] = b2s[16*m + 8*hi + j];
    #pragma unroll
    for (int k = 0; k < 4; ++k){
      v16h a = load_a(w2s, W2RS, 16*m, 32*k, lane);
      acc = __builtin_amdgcn_wmma_f32_16x16x32_f16(false, a, false, bh[k], (short)0, acc, false, false);
    }
    const int pos = pbase + wave*16 + col;
    #pragma unroll
    for (int j = 0; j < 8; ++j){
      int och = 16*m + 8*hi + j;          /* z1_new channel i -> out channel 2i */
      int g = nbase + (2*och)*P_ + pos;
      __builtin_nontemporal_store(acc[j] + z0[g], out + g);
    }
  }

  /* odd out channels: out[2i+1] = norm(z0[64+i]) + z0[2i+1] (float4 streams) */
  #pragma unroll
  for (int i = 0; i < 8; ++i){
    int u  = t + i*256;
    int ch = u >> 5;
    int p4 = (u & 31) * 4;
    const f4 z2v = *(const f4*)(z0 + nbase + (CC_ + ch)*P_ + pbase + p4);
    const f4 rv  = *(const f4*)(z0 + nbase + (2*ch + 1)*P_ + pbase + p4);
    f4 o4 = (z2v - mu) * rs + rv;
    __builtin_nontemporal_store(o4, (f4*)(out + nbase + (2*ch + 1)*P_ + pbase + p4));
  }
}

extern "C" void kernel_launch(void* const* d_in, const int* in_sizes, int n_in,
                              void* d_out, int out_size, void* d_ws, size_t ws_size,
                              hipStream_t stream)
{
  const float* z0 = (const float*)d_in[0];
  const float* w1 = (const float*)d_in[1];
  const float* b1 = (const float*)d_in[2];
  const float* w2 = (const float*)d_in[3];
  const float* b2 = (const float*)d_in[4];
  float* out = (float*)d_out;

  float2* partials = (float2*)d_ws;               /* 16*288 float2 */
  float2* musig    = partials + N_*SBLK_;         /* 16 float2     */

  cp_stats1<<<dim3(SBLK_, N_), 256, 0, stream>>>(z0, partials);
  cp_stats2<<<N_, 256, 0, stream>>>(partials, musig);
  cp_main  <<<dim3(P_/PTILE_, N_), 256, 0, stream>>>(z0, w1, b1, w2, b2, musig, out);
}
